// MI_RIM_10471130268051
// MI455X (gfx1250) — matrix operand order repaired and checked by reading, not compiled, not run on hardware
//
#include <hip/hip_runtime.h>
#include <stdint.h>

// ---------------------------------------------------------------------------
// MI-RIM recurrent scan for MI455X (gfx1250).
//   - weights converted fp32->bf16 once per launch into d_ws (104MB, fits the
//     192MB L2 -> all 512 steps stream weights from L2, not HBM)
//   - persistent kernel: 16 workgroups (one per module), 512 thr (16 waves)
//     => generous VGPR budget, no spills, 4 waves/SIMD for latency hiding
//   - GEMV via V_WMMA_F32_16X16X32_BF16 (vector broadcast into all B columns)
//     with an explicit depth-2 software pipeline (2 A/B fragment sets) so
//     each wave keeps ~4 b128 weight loads in flight
//   - 2 device-wide sense-reversing barriers per timestep for cross-module
//     top-k mask + 16x16 attention mixing
// Required ws_size >= ~110 MB.
// ---------------------------------------------------------------------------

#define DEVINL __device__ __forceinline__

constexpr int M_ = 16, T_ = 512, D_ = 512, H_ = 512, QK_ = 256, G_ = 2048;
constexpr int NWG    = 16;
constexpr int NTHR   = 512;
constexpr int NWAVES = NTHR / 32;

// d_ws layout (bytes)
constexpr size_t WS_CNT  = 0;                 // 2 x u32 barrier state
constexpr size_t WS_ATT0 = 64;                // 16 f32
constexpr size_t WS_K2   = 256;               // 16*512 f32
constexpr size_t WS_V2   = 256 + 32768;       // 16*512 f32
constexpr size_t WS_WOFF = 131072;            // bf16 (ushort) weight region

// offsets inside weight region (ushort elements)
constexpr size_t OFF_WQ  = 0;
constexpr size_t OFF_WK  = OFF_WQ  + (size_t)M_ * QK_ * D_;
constexpr size_t OFF_WV  = OFF_WK  + (size_t)M_ * QK_ * D_;
constexpr size_t OFF_WIH = OFF_WV  + (size_t)M_ * D_  * D_;
constexpr size_t OFF_WHH = OFF_WIH + (size_t)M_ * G_  * D_;
constexpr size_t OFF_WQG = OFF_WHH + (size_t)M_ * G_  * H_;   // transposed (O,H)
constexpr size_t OFF_WKG = OFF_WQG + (size_t)M_ * H_  * H_;
constexpr size_t OFF_WVG = OFF_WKG + (size_t)M_ * H_  * H_;

constexpr size_t OUT_IDX = (size_t)T_ * M_ * H_;  // all_idx after all_h

typedef __attribute__((ext_vector_type(16))) __bf16 v16bf;
typedef __attribute__((ext_vector_type(8)))  float  v8f;

union Frag { v16bf v; uint4 u[2]; };

DEVINL unsigned short f2b(float f) {          // fp32 -> bf16, round-nearest-even
  unsigned u = __float_as_uint(f);
  unsigned r = 0x7FFFu + ((u >> 16) & 1u);
  return (unsigned short)((u + r) >> 16);
}
DEVINL float sigm(float x) { return 1.f / (1.f + __expf(-x)); }

// ---------------------------------------------------------------------------
// GEMV tile: accumulate  y[g0..g0+15] += W[g0:g0+16, 0:dlen] * v   via WMMA.
// A = 16x32 weight tile (row-major, ld=512, bf16) loaded per the gfx1250
// 16-bit A layout (lanes 0-15: K0..7,K16..23 ; lanes 16-31: K8..15,K24..31).
// B = vector chunk broadcast into all 16 columns (lanes 0-15: K0..7,K16..23 ;
// lanes 16-31: K8..15,K24..31), so every column of D holds the valid result.
// Depth-2 software pipeline: chunk c+1 loads issue before the WMMA on chunk c.
// ---------------------------------------------------------------------------
DEVINL v8f gemv_mat(v8f acc, const unsigned short* __restrict__ W, int g0,
                    const unsigned short* __restrict__ vb, int dlen, int lane) {
  const int row   = g0 + (lane & 15);
  const int half8 = (lane >> 4) << 3;   // 0 or 8  (A K-offset)
  const int bsel  = (lane >> 4) << 3;   // 0 or 8  (B K-offset)
  const uint4* __restrict__ ap =
      (const uint4*)(W + (size_t)row * 512 + half8);   // ap[4c],ap[4c+2] = chunk c
  const unsigned short* vbl = vb + bsel;
  const int nc = dlen >> 5;                            // 32-wide K chunks
  __builtin_prefetch(W + (size_t)(row + 16) * 512, 0, 1);  // next tile row

  Frag a0, a1, b0, b1;
  a0.u[0] = ap[0];
  a0.u[1] = ap[2];
  {
    const uint4* bp = (const uint4*)vbl;
    b0.u[0] = bp[0];
    b0.u[1] = bp[2];
  }
#pragma unroll
  for (int c = 0; c < nc; c += 2) {
    // stage in chunk c+1 while chunk c computes
    a1.u[0] = ap[4 * c + 4];
    a1.u[1] = ap[4 * c + 6];
    {
      const uint4* bp = (const uint4*)(vbl + 32 * (c + 1));
      b1.u[0] = bp[0];
      b1.u[1] = bp[2];
    }
    acc = __builtin_amdgcn_wmma_f32_16x16x32_bf16(
        false, a0.v, false, b0.v, (short)0, acc, false, false);
    if (c + 2 < nc) {  // stage in chunk c+2 while chunk c+1 computes
      a0.u[0] = ap[4 * c + 8];
      a0.u[1] = ap[4 * c + 10];
      const uint4* bp = (const uint4*)(vbl + 32 * (c + 2));
      b0.u[0] = bp[0];
      b0.u[1] = bp[2];
    }
    acc = __builtin_amdgcn_wmma_f32_16x16x32_bf16(
        false, a1.v, false, b1.v, (short)0, acc, false, false);
  }
  return acc;
}

// C/D layout: lane 0 VGPR r -> row g0+r ; lane 16 VGPR r -> row g0+8+r.
DEVINL void store_tile(float* dst, int g0, v8f acc, int lane,
                       const float* b0, const float* b1) {
  if ((lane & 15) == 0) {
    int base = g0 + ((lane >> 4) << 3);
#pragma unroll
    for (int r = 0; r < 8; ++r) {
      float y = acc[r];
      if (b0) y += b0[base + r];
      if (b1) y += b1[base + r];
      dst[base + r] = y;
    }
  }
}

// ---------------------------------------------------------------------------
// Device-wide sense-reversing barrier across the 16 persistent workgroups.
// ---------------------------------------------------------------------------
DEVINL void grid_barrier(unsigned* cnt, unsigned* gen) {
  __builtin_amdgcn_fence(__ATOMIC_RELEASE, "agent");
  __syncthreads();
  if (threadIdx.x == 0) {
    unsigned g = __hip_atomic_load(gen, __ATOMIC_RELAXED, __HIP_MEMORY_SCOPE_AGENT);
    if (__hip_atomic_fetch_add(cnt, 1u, __ATOMIC_ACQ_REL,
                               __HIP_MEMORY_SCOPE_AGENT) == NWG - 1) {
      __hip_atomic_store(cnt, 0u, __ATOMIC_RELAXED, __HIP_MEMORY_SCOPE_AGENT);
      __hip_atomic_fetch_add(gen, 1u, __ATOMIC_RELEASE, __HIP_MEMORY_SCOPE_AGENT);
    } else {
      while (__hip_atomic_load(gen, __ATOMIC_ACQUIRE,
                               __HIP_MEMORY_SCOPE_AGENT) == g)
        __builtin_amdgcn_s_sleep(2);
    }
  }
  __syncthreads();
  __builtin_amdgcn_fence(__ATOMIC_ACQUIRE, "agent");
}

// ---------------------------------------------------------------------------
// Prep kernels
// ---------------------------------------------------------------------------
__global__ void k_init(unsigned* cnt) {
  if (threadIdx.x == 0) { cnt[0] = 0u; cnt[1] = 0u; }
}

__global__ void k_convert(const float* __restrict__ src,
                          unsigned short* __restrict__ dst, int n) {
  int stride = gridDim.x * blockDim.x;
  for (int i = blockIdx.x * blockDim.x + threadIdx.x; i < n; i += stride)
    dst[i] = f2b(src[i]);
}

// (M,H,O) fp32 -> (M,O,H) bf16  (contraction index first -> repack transposed)
__global__ void k_transpose_g(const float* __restrict__ src,
                              unsigned short* __restrict__ dst) {
  const size_t total = (size_t)M_ * H_ * H_;
  size_t stride = (size_t)gridDim.x * blockDim.x;
  for (size_t i = (size_t)blockIdx.x * blockDim.x + threadIdx.x; i < total;
       i += stride) {
    size_t m = i >> 18;          // /(512*512)
    size_t rem = i & 0x3FFFF;
    size_t o = rem >> 9, h = rem & 511;
    dst[i] = f2b(src[(m << 18) + (h << 9) + o]);
  }
}

// ---------------------------------------------------------------------------
// Main persistent kernel: 16 WGs x 512 threads, loops over all T steps.
// ---------------------------------------------------------------------------
struct __align__(32) Smem {
  unsigned short xb[D_], hb[H_], selb[D_], thb[H_];  // bf16 activation vectors
  float hs[H_], cs[H_], ths[H_], q2s[H_];
  float qs[QK_], kxs[QK_], val1s[D_];
  float gs[G_];
  float red0[QK_], red1[QK_];
  float smask;
};

__global__ __launch_bounds__(NTHR) void k_main(
    const float* __restrict__ inp,  const float* __restrict__ bq,
    const float* __restrict__ bk,   const float* __restrict__ bv,
    const float* __restrict__ bih,  const float* __restrict__ bhh,
    void* wsv, float* __restrict__ out) {
  __shared__ Smem sm;

  unsigned char* ws = (unsigned char*)wsv;
  unsigned* cnt = (unsigned*)(ws + WS_CNT);
  unsigned* gen = cnt + 1;
  float* att0buf = (float*)(ws + WS_ATT0);
  float* k2buf   = (float*)(ws + WS_K2);
  float* v2buf   = (float*)(ws + WS_V2);
  const unsigned short* wb = (const unsigned short*)(ws + WS_WOFF);

  const int m    = blockIdx.x;
  const int tid  = threadIdx.x;
  const int lane = tid & 31;
  const int w    = tid >> 5;

  const unsigned short* Wq_m  = wb + OFF_WQ  + (size_t)m * QK_ * D_;
  const unsigned short* Wk_m  = wb + OFF_WK  + (size_t)m * QK_ * D_;
  const unsigned short* Wv_m  = wb + OFF_WV  + (size_t)m * D_  * D_;
  const unsigned short* Wih_m = wb + OFF_WIH + (size_t)m * G_  * D_;
  const unsigned short* Whh_m = wb + OFF_WHH + (size_t)m * G_  * H_;
  const unsigned short* WqgT  = wb + OFF_WQG + (size_t)m * H_  * H_;
  const unsigned short* WkgT  = wb + OFF_WKG + (size_t)m * H_  * H_;
  const unsigned short* WvgT  = wb + OFF_WVG + (size_t)m * H_  * H_;
  const float* bq_m  = bq  + (size_t)m * QK_;
  const float* bk_m  = bk  + (size_t)m * QK_;
  const float* bv_m  = bv  + (size_t)m * D_;
  const float* bih_m = bih + (size_t)m * G_;
  const float* bhh_m = bhh + (size_t)m * G_;

  // init state
  for (int i = tid; i < H_; i += NTHR) {
    sm.hs[i] = 0.f; sm.cs[i] = 0.f; sm.hb[i] = f2b(0.f);
  }
  __syncthreads();

  const float scale = 1.0f / 16.0f;  // 1/sqrt(QK)

  for (int t = 0; t < T_; ++t) {
    // ---- load x_t (bf16 copy only; fp32 x is not needed elsewhere) ----
    for (int i = tid; i < D_; i += NTHR)
      sm.xb[i] = f2b(inp[((size_t)m * T_ + t) * D_ + i]);
    __syncthreads();

    // ---- phase 1: q = Wq h + bq ; key_x = Wk x + bk ; val1 = Wv x + bv ----
    // 64 tiles total: [0,16)=q, [16,32)=key_x, [32,64)=val1
    for (int tile = w; tile < 64; tile += NWAVES) {
      v8f acc = {0.f,0.f,0.f,0.f,0.f,0.f,0.f,0.f};
      if (tile < 16) {
        acc = gemv_mat(acc, Wq_m, tile * 16, sm.hb, D_, lane);
        store_tile(sm.qs, tile * 16, acc, lane, bq_m, nullptr);
      } else if (tile < 32) {
        acc = gemv_mat(acc, Wk_m, (tile - 16) * 16, sm.xb, D_, lane);
        store_tile(sm.kxs, (tile - 16) * 16, acc, lane, bk_m, nullptr);
      } else {
        acc = gemv_mat(acc, Wv_m, (tile - 32) * 16, sm.xb, D_, lane);
        store_tile(sm.val1s, (tile - 32) * 16, acc, lane, bv_m, nullptr);
      }
    }
    __syncthreads();

    // logits + softmax(2)
    if (tid < QK_) {
      sm.red0[tid] = sm.qs[tid] * bk_m[tid];
      sm.red1[tid] = sm.qs[tid] * sm.kxs[tid];
    }
    __syncthreads();
    for (int s = QK_ / 2; s > 0; s >>= 1) {
      if (tid < s) {
        sm.red0[tid] += sm.red0[tid + s];
        sm.red1[tid] += sm.red1[tid + s];
      }
      __syncthreads();
    }
    if (tid == 0) {
      float l0 = sm.red0[0] * scale, l1 = sm.red1[0] * scale;
      float mx = fmaxf(l0, l1);
      float e0 = __expf(l0 - mx), e1 = __expf(l1 - mx);
      float a0 = e0 / (e0 + e1);
      sm.red0[0] = a0;
      sm.red1[0] = 1.f - a0;
      att0buf[m] = a0;  // publish for global top-k
    }
    __syncthreads();
    {
      float a0 = sm.red0[0], a1 = sm.red1[0];
      for (int i = tid; i < D_; i += NTHR)
        sm.selb[i] = f2b(a0 * bv_m[i] + a1 * sm.val1s[i]);
    }
    grid_barrier(cnt, gen);  // ---- all att0 published ----

    // ---- top-4 smallest att0 (== jax top_k(-att0,4), stable) ----
    if (tid == 0) {
      float a[16]; bool tk[16];
      for (int j = 0; j < 16; ++j) { a[j] = att0buf[j]; tk[j] = false; }
      float mk = 0.f;
      for (int k = 0; k < 4; ++k) {
        int bi = 0; float bvv = 3.4e38f;
        for (int j = 0; j < 16; ++j)
          if (!tk[j] && a[j] < bvv) { bvv = a[j]; bi = j; }
        tk[bi] = true;
        if (bi == m) mk = 1.f;
        if (m == 0) ((int*)out)[OUT_IDX + (size_t)t * 4 + k] = bi;
      }
      sm.smask = mk;
    }
    __syncthreads();
    const float mb = sm.smask;

    // ---- phase 2: gates = W_ih sel + b_ih + W_hh h + b_hh ; LSTM cell ----
    for (int tile = w; tile < G_ / 16; tile += NWAVES) {
      v8f acc = {0.f,0.f,0.f,0.f,0.f,0.f,0.f,0.f};
      acc = gemv_mat(acc, Wih_m, tile * 16, sm.selb, D_, lane);
      acc = gemv_mat(acc, Whh_m, tile * 16, sm.hb,  H_, lane);
      store_tile(sm.gs, tile * 16, acc, lane, bih_m, bhh_m);
    }
    __syncthreads();
    if (tid < H_) {
      float i_ = sigm(sm.gs[tid]);
      float f_ = sigm(sm.gs[H_ + tid]);
      float g_ = tanhf(sm.gs[2 * H_ + tid]);
      float o_ = sigm(sm.gs[3 * H_ + tid]);
      float cn = f_ * sm.cs[tid] + i_ * g_;
      float hn = o_ * tanhf(cn);
      float th = mb * hn + (1.f - mb) * sm.hs[tid];
      sm.cs[tid] = mb * cn;           // new_c = m * c_new
      sm.ths[tid] = th;
      sm.thb[tid] = f2b(th);
    }
    __syncthreads();

    // ---- phase 3: q2/k2/v2 = Wg^T temp_h (pre-transposed weights) ----
    for (int tile = w; tile < 96; tile += NWAVES) {
      const unsigned short* Wg = (tile < 32) ? WqgT : (tile < 64 ? WkgT : WvgT);
      int tt = tile & 31;
      v8f acc = {0.f,0.f,0.f,0.f,0.f,0.f,0.f,0.f};
      acc = gemv_mat(acc, Wg, tt * 16, sm.thb, H_, lane);
      float* dst = (tile < 32) ? sm.q2s
                 : (tile < 64 ? (k2buf + (size_t)m * H_)
                              : (v2buf + (size_t)m * H_));
      store_tile(dst, tt * 16, acc, lane, nullptr, nullptr);
    }
    grid_barrier(cnt, gen);  // ---- all k2/v2 published ----

    // ---- phase 4: att2 row m, inter, new_h ----
    if (w < 16) {  // wave j: dot(q2, k2[j])
      const float* k2j = k2buf + (size_t)w * H_;
      float s = 0.f;
      for (int i = lane; i < H_; i += 32) s += sm.q2s[i] * k2j[i];
      for (int off = 16; off > 0; off >>= 1) s += __shfl_down(s, off, 32);
      if (lane == 0) sm.red0[w] = s;
    }
    __syncthreads();
    if (tid == 0) {
      float mx = -3.4e38f;
      for (int j = 0; j < 16; ++j) mx = fmaxf(mx, sm.red0[j]);
      float sum = 0.f;
      for (int j = 0; j < 16; ++j) {
        float p = __expf(sm.red0[j] - mx);
        sm.red1[j] = p; sum += p;
      }
      float inv = 1.f / sum;
      for (int j = 0; j < 16; ++j) sm.red1[j] *= inv;
    }
    __syncthreads();
    if (tid < H_) {
      float acc = 0.f;
      for (int j = 0; j < 16; ++j)
        acc += sm.red1[j] * v2buf[(size_t)j * H_ + tid];
      float nh = mb * acc + (1.f - mb) * sm.hs[tid] + mb * sm.ths[tid];
      sm.hs[tid] = nh;
      sm.hb[tid] = f2b(nh);
      out[(size_t)t * (M_ * H_) + (size_t)m * H_ + tid] = nh;
    }
    __syncthreads();
  }
}

// ---------------------------------------------------------------------------
extern "C" void kernel_launch(void* const* d_in, const int* in_sizes, int n_in,
                              void* d_out, int out_size, void* d_ws,
                              size_t ws_size, hipStream_t stream) {
  (void)in_sizes; (void)n_in; (void)out_size; (void)ws_size;
  const float* inp = (const float*)d_in[0];
  const float* Wq  = (const float*)d_in[1];
  const float* bq  = (const float*)d_in[2];
  const float* Wk  = (const float*)d_in[3];
  const float* bk  = (const float*)d_in[4];
  const float* Wv  = (const float*)d_in[5];
  const float* bv  = (const float*)d_in[6];
  const float* Wih = (const float*)d_in[7];
  const float* bih = (const float*)d_in[8];
  const float* Whh = (const float*)d_in[9];
  const float* bhh = (const float*)d_in[10];
  const float* Wqg = (const float*)d_in[11];
  const float* Wkg = (const float*)d_in[12];
  const float* Wvg = (const float*)d_in[13];

  unsigned char* ws = (unsigned char*)d_ws;
  unsigned short* wb = (unsigned short*)(ws + WS_WOFF);

  k_init<<<1, 64, 0, stream>>>((unsigned*)ws);

  k_convert<<<4096, 256, 0, stream>>>(Wq,  wb + OFF_WQ,  M_ * QK_ * D_);
  k_convert<<<4096, 256, 0, stream>>>(Wk,  wb + OFF_WK,  M_ * QK_ * D_);
  k_convert<<<4096, 256, 0, stream>>>(Wv,  wb + OFF_WV,  M_ * D_  * D_);
  k_convert<<<4096, 256, 0, stream>>>(Wih, wb + OFF_WIH, M_ * G_  * D_);
  k_convert<<<4096, 256, 0, stream>>>(Whh, wb + OFF_WHH, M_ * G_  * H_);
  k_transpose_g<<<4096, 256, 0, stream>>>(Wqg, wb + OFF_WQG);
  k_transpose_g<<<4096, 256, 0, stream>>>(Wkg, wb + OFF_WKG);
  k_transpose_g<<<4096, 256, 0, stream>>>(Wvg, wb + OFF_WVG);

  k_main<<<NWG, NTHR, 0, stream>>>(inp, bq, bk, bv, bih, bhh, d_ws,
                                   (float*)d_out);
}
